// RWKVRegionCell_65558380806209
// MI455X (gfx1250) — matrix operand order repaired and checked by reading, not compiled, not run on hardware
//
#include <hip/hip_runtime.h>
#include <hip/hip_bf16.h>

// ---------------------------------------------------------------------------
// RWKV region cell for MI455X (gfx1250), decomposed into:
//   1) rmsnorm + bf16 hi/lo split of xn            (memory bound, trivial)
//   2) 3x bf16x2-split WMMA GEMMs -> r,k,v         (compute bound, v_wmma)
//   3) sequential EMA scan, parallel over B*d      (latency chain, cheap)
//   4) bf16x2 WMMA GEMM  h = xn + y@Wo.T           (compute bound, v_wmma)
//   5) exact KWTA top-128 threshold per row        (bitwise rank search)
// Precision: split-bf16 (hi+lo) with fp32 accumulation ~ fp32 GEMM accuracy
// at bf16 WMMA throughput (3 wmma per logical fp32 MAC tile).
// ---------------------------------------------------------------------------

typedef __attribute__((ext_vector_type(16))) __bf16 v16bf;
typedef __attribute__((ext_vector_type(8)))  float  v8f;

using u32 = unsigned int;
using u16 = unsigned short;

#define RB  8
#define RT  4096
#define RD  1024
#define RN  (RB * RT)          // 32768 rows
#define KWTA_K 128
#define M_TIME 16

union FragBf {
    v16bf v;
    uint4 q[2];
};

// ---------------- bf16 split helpers -----------------------------------
__device__ __forceinline__ u16 f2bf_rne(float f) {
    u32 u = __float_as_uint(f);
    u32 r = u + 0x7FFFu + ((u >> 16) & 1u);
    return (u16)(r >> 16);
}
__device__ __forceinline__ float bf2f(u16 h) {
    return __uint_as_float(((u32)h) << 16);
}
__device__ __forceinline__ void split_bf(float f, u16& hi, u16& lo) {
    hi = f2bf_rne(f);
    lo = f2bf_rne(f - bf2f(hi));
}

// ---------------- 1) weight split ---------------------------------------
__global__ __launch_bounds__(256) void k_split_f32(const float* __restrict__ W,
                                                   u16* __restrict__ Wh,
                                                   u16* __restrict__ Wl, int n) {
    int i = blockIdx.x * 256 + threadIdx.x;
    if (i < n) {
        u16 h, l;
        split_bf(W[i], h, l);
        Wh[i] = h;
        Wl[i] = l;
    }
}

// ---------------- 2) rmsnorm + split ------------------------------------
__global__ __launch_bounds__(256) void k_rmsnorm_split(const float* __restrict__ x,
                                                       const float* __restrict__ gamma,
                                                       float* __restrict__ xn,
                                                       u16* __restrict__ xh,
                                                       u16* __restrict__ xl) {
    __shared__ float red[256];
    const int tid = threadIdx.x;
    const size_t base = (size_t)blockIdx.x * RD;
    float vals[4];
    float ss = 0.f;
#pragma unroll
    for (int i = 0; i < 4; ++i) {
        vals[i] = x[base + tid + i * 256];
        ss += vals[i] * vals[i];
    }
    red[tid] = ss;
    __syncthreads();
    for (int s = 128; s > 0; s >>= 1) {
        if (tid < s) red[tid] += red[tid + s];
        __syncthreads();
    }
    const float inv = rsqrtf(red[0] * (1.0f / RD) + 1e-6f);
#pragma unroll
    for (int i = 0; i < 4; ++i) {
        const int c = tid + i * 256;
        const float v = vals[i] * inv * gamma[c];
        xn[base + c] = v;
        u16 h, l;
        split_bf(v, h, l);
        xh[base + c] = h;
        xl[base + c] = l;
    }
}

// ---------------- 3) bf16x2 split WMMA GEMM -----------------------------
// Out[M,N] = (Ah+Al)[M,K] * (Bh+Bl)^T[K,N]  (+ residual), fp32 accumulate.
// B operand is W row-major [n,k] -> column n of the K x N matrix is W row n,
// contiguous in k, which matches the wave32 B-fragment layout directly.
// WG: 256 thr = 8 waves (4 along M x 2 along N). Wave tile 32x32 = 2x2 wmma.
__global__ __launch_bounds__(256, 1) void k_gemm_bf16x2(
    const u16* __restrict__ Ah, const u16* __restrict__ Al,
    const u16* __restrict__ Bh, const u16* __restrict__ Bl,
    const float* __restrict__ residual, float* __restrict__ Out,
    int K, int Ncols) {
    const int lane  = threadIdx.x & 31;
    const int wave  = threadIdx.x >> 5;
    const int waveM = wave & 3;
    const int waveN = wave >> 2;
    const int rowBase = blockIdx.y * 128 + waveM * 32;
    const int colBase = blockIdx.x * 64 + waveN * 32;
    const int m    = lane & 15;
    const int half = lane >> 4;
    // A fragment (16-bit 16x32, ISA 7.12.2): lanes 0-15 hold K 0..7 & 16..23,
    // lanes 16-31 hold K 8..15 & 24..31, row M = lane&15.
    const int aK0 = half ? 8 : 0;
    const int aK1 = half ? 24 : 16;
    // B fragment: lane column = lane&15; lanes 0-15 K 0..7 & 16..23, lanes 16-31 K 8..15 & 24..31.
    const int bK0 = half ? 8 : 0;

    v8f acc[2][2] = {};

    for (int k0 = 0; k0 < K; k0 += 32) {
        FragBf ah[2], al[2], bh[2], bl[2];
#pragma unroll
        for (int mt = 0; mt < 2; ++mt) {
            const size_t ro = (size_t)(rowBase + mt * 16 + m) * K + k0;
            ah[mt].q[0] = *(const uint4*)(Ah + ro + aK0);
            ah[mt].q[1] = *(const uint4*)(Ah + ro + aK1);
            al[mt].q[0] = *(const uint4*)(Al + ro + aK0);
            al[mt].q[1] = *(const uint4*)(Al + ro + aK1);
        }
#pragma unroll
        for (int nt = 0; nt < 2; ++nt) {
            const size_t co = (size_t)(colBase + nt * 16 + m) * K + k0 + bK0;
            bh[nt].q[0] = *(const uint4*)(Bh + co);
            bh[nt].q[1] = *(const uint4*)(Bh + co + 16);
            bl[nt].q[0] = *(const uint4*)(Bl + co);
            bl[nt].q[1] = *(const uint4*)(Bl + co + 16);
        }
#pragma unroll
        for (int mt = 0; mt < 2; ++mt) {
#pragma unroll
            for (int nt = 0; nt < 2; ++nt) {
                // hi*hi + hi*lo + lo*hi  (drop lo*lo: ~2^-16 relative)
                acc[mt][nt] = __builtin_amdgcn_wmma_f32_16x16x32_bf16(
                    false, ah[mt].v, false, bh[nt].v, (short)0, acc[mt][nt], false, false);
                acc[mt][nt] = __builtin_amdgcn_wmma_f32_16x16x32_bf16(
                    false, ah[mt].v, false, bl[nt].v, (short)0, acc[mt][nt], false, false);
                acc[mt][nt] = __builtin_amdgcn_wmma_f32_16x16x32_bf16(
                    false, al[mt].v, false, bh[nt].v, (short)0, acc[mt][nt], false, false);
            }
        }
    }

    // D layout (f32 16x16): lane 0-15 -> N=lane, M=r; lane 16-31 -> N=lane-16, M=8+r
#pragma unroll
    for (int mt = 0; mt < 2; ++mt) {
#pragma unroll
        for (int nt = 0; nt < 2; ++nt) {
#pragma unroll
            for (int r = 0; r < 8; ++r) {
                const int row = rowBase + mt * 16 + r + (half ? 8 : 0);
                const int col = colBase + nt * 16 + m;
                const size_t idx = (size_t)row * Ncols + col;
                float v = acc[mt][nt][r];
                if (residual) v += residual[idx];
                Out[idx] = v;
            }
        }
    }
}

// ---------------- 4) recurrent EMA scan ---------------------------------
// One thread per (b, channel). Sequential over T; applies sigmoid(r), exp(k)
// and RoPE on the first 2*M_TIME channels of v; writes bf16 hi/lo split of y.
__global__ __launch_bounds__(256) void k_scan(const float* __restrict__ R,
                                              const float* __restrict__ Kp,
                                              const float* __restrict__ V,
                                              const float* __restrict__ pos,
                                              const float* __restrict__ raw_decay,
                                              u16* __restrict__ yh,
                                              u16* __restrict__ yl) {
    const int b = blockIdx.x >> 2;
    const int c = ((blockIdx.x & 3) << 8) + threadIdx.x;
    const float rd  = raw_decay[c];
    const float lam = expf(-log1pf(expf(rd)));   // exp(-softplus(rd))

    const bool rope = (c < 2 * M_TIME);
    float freq = 0.f;
    int pairBase = 0;
    bool oddc = false;
    if (rope) {
        const int i = c >> 1;
        freq = expf(-((float)i / (float)M_TIME) * logf(10000.0f));
        pairBase = c & ~1;
        oddc = (c & 1) != 0;
    }

    float num = 0.f, den = 0.f;
    const size_t rowStart = (size_t)b * RT * RD;
    for (int t = 0; t < RT; ++t) {
        const size_t rowOff = rowStart + (size_t)t * RD;
        const size_t idx = rowOff + c;
        const float rg = R[idx];
        const float r  = 1.f / (1.f + expf(-rg));
        const float w  = expf(Kp[idx]);
        float v;
        if (rope) {
            const float ve = V[rowOff + pairBase];
            const float vo = V[rowOff + pairBase + 1];
            const float th = pos[t] * freq;
            const float ct = cosf(th), st = sinf(th);
            v = oddc ? (ve * st + vo * ct) : (ve * ct - vo * st);
        } else {
            v = V[idx];
        }
        num = num * lam + w * v;
        den = den * lam + w;
        const float y = r * (num / (den + 1e-9f));
        u16 h, l;
        split_bf(y, h, l);
        yh[idx] = h;
        yl[idx] = l;
    }
}

// ---------------- 5) exact KWTA ------------------------------------------
// Order-preserving float->uint key; bitwise binary search for the k-th
// largest key (max t with count(key>=t) >= k). Matches reference tie rule.
__device__ __forceinline__ u32 fkey(float f) {
    u32 u = __float_as_uint(f);
    return (u & 0x80000000u) ? ~u : (u | 0x80000000u);
}

__global__ __launch_bounds__(256) void k_kwta(float* __restrict__ H) {
    __shared__ u32 cnt[256];
    const int tid = threadIdx.x;
    const size_t base = (size_t)blockIdx.x * RD;
    float vals[4];
    u32 keys[4];
#pragma unroll
    for (int i = 0; i < 4; ++i) {
        vals[i] = H[base + tid + i * 256];
        keys[i] = fkey(vals[i]);
    }
    u32 ans = 0;
    for (int bit = 31; bit >= 0; --bit) {
        const u32 cand = ans | (1u << bit);
        u32 c = 0;
#pragma unroll
        for (int i = 0; i < 4; ++i) c += (keys[i] >= cand) ? 1u : 0u;
        cnt[tid] = c;
        __syncthreads();
        for (int s = 128; s > 0; s >>= 1) {
            if (tid < s) cnt[tid] += cnt[tid + s];
            __syncthreads();
        }
        const u32 total = cnt[0];
        __syncthreads();
        if (total >= (u32)KWTA_K) ans = cand;
    }
#pragma unroll
    for (int i = 0; i < 4; ++i) {
        H[base + tid + i * 256] = (keys[i] >= ans) ? vals[i] : 0.0f;
    }
}

// ---------------- host orchestration -------------------------------------
static inline size_t align256(size_t x) { return (x + 255) & ~(size_t)255; }

extern "C" void kernel_launch(void* const* d_in, const int* in_sizes, int n_in,
                              void* d_out, int out_size, void* d_ws, size_t ws_size,
                              hipStream_t stream) {
    (void)in_sizes; (void)n_in; (void)out_size;
    const float* x         = (const float*)d_in[0];
    const float* pos       = (const float*)d_in[1];
    const float* gamma     = (const float*)d_in[2];
    const float* Wr        = (const float*)d_in[3];
    const float* Wk        = (const float*)d_in[4];
    const float* Wv        = (const float*)d_in[5];
    const float* Wo        = (const float*)d_in[6];
    const float* raw_decay = (const float*)d_in[7];
    float* out = (float*)d_out;

    char* ws = (char*)d_ws;
    size_t off = 0;
    auto carve = [&](size_t bytes) {
        char* p = ws + off;
        off = align256(off + bytes);
        return p;
    };
    const size_t nElems = (size_t)RN * RD;          // 32768 * 1024
    float* xn  = (float*)carve(nElems * 4);
    u16*   xh  = (u16*)  carve(nElems * 2);         // reused as y_hi after scan
    u16*   xl  = (u16*)  carve(nElems * 2);         // reused as y_lo after scan
    float* Rb  = (float*)carve(nElems * 4);
    float* Kb  = (float*)carve(nElems * 4);
    float* Vb  = (float*)carve(nElems * 4);
    const size_t wElems = (size_t)RD * RD;
    u16* Wrh = (u16*)carve(wElems * 2);
    u16* Wrl = (u16*)carve(wElems * 2);
    u16* Wkh = (u16*)carve(wElems * 2);
    u16* Wkl = (u16*)carve(wElems * 2);
    u16* Wvh = (u16*)carve(wElems * 2);
    u16* Wvl = (u16*)carve(wElems * 2);
    u16* Woh = (u16*)carve(wElems * 2);
    u16* Wol = (u16*)carve(wElems * 2);
    (void)ws_size;  // layout requires ~657 MiB of scratch

    // 1) split the four weight matrices into bf16 hi/lo
    const int wBlocks = (int)((wElems + 255) / 256);
    k_split_f32<<<wBlocks, 256, 0, stream>>>(Wr, Wrh, Wrl, (int)wElems);
    k_split_f32<<<wBlocks, 256, 0, stream>>>(Wk, Wkh, Wkl, (int)wElems);
    k_split_f32<<<wBlocks, 256, 0, stream>>>(Wv, Wvh, Wvl, (int)wElems);
    k_split_f32<<<wBlocks, 256, 0, stream>>>(Wo, Woh, Wol, (int)wElems);

    // 2) rmsnorm all rows + split xn
    k_rmsnorm_split<<<RN, 256, 0, stream>>>(x, gamma, xn, xh, xl);

    // 3) projections r,k,v : [32768,1024] x [1024,1024]^T
    dim3 gGrid(RD / 64, RN / 128);  // 16 x 256
    k_gemm_bf16x2<<<gGrid, 256, 0, stream>>>(xh, xl, Wrh, Wrl, nullptr, Rb, RD, RD);
    k_gemm_bf16x2<<<gGrid, 256, 0, stream>>>(xh, xl, Wkh, Wkl, nullptr, Kb, RD, RD);
    k_gemm_bf16x2<<<gGrid, 256, 0, stream>>>(xh, xl, Wvh, Wvl, nullptr, Vb, RD, RD);

    // 4) recurrent scan (writes y split into xh/xl, which are dead by now)
    k_scan<<<(RB * RD) / 256, 256, 0, stream>>>(Rb, Kb, Vb, pos, raw_decay, xh, xl);

    // 5) h = xn + y @ Wo.T  -> d_out
    k_gemm_bf16x2<<<gGrid, 256, 0, stream>>>(xh, xl, Woh, Wol, xn, out, RD, RD);

    // 6) exact KWTA (top-128 threshold) in place on d_out
    k_kwta<<<RN, 256, 0, stream>>>(out);
}
